// LePEAttention_30588757082812
// MI455X (gfx1250) — matrix operand order repaired and checked by reading, not compiled, not run on hardware
//
#include <hip/hip_runtime.h>

typedef _Float16 f16;
typedef __attribute__((ext_vector_type(16))) _Float16 v16h;
typedef __attribute__((ext_vector_type(8)))  _Float16 h8;
typedef __attribute__((ext_vector_type(2)))  _Float16 h2;
typedef __attribute__((ext_vector_type(8)))  float    v8f;
typedef __attribute__((ext_vector_type(4)))  float    f4;

#define DIM   256
#define RES   32
#define HEADS 8
#define HD    32          // head dim
#define SWIN  256         // tokens per window = 32*8
#define BATCH 2
#define LTOK  (RES*RES*RES)
#define SCALE 0.1767766952966369f   // 32^-0.5
#define SH_LD 264                    // f16 score row stride (16B aligned, bank skew)

// ---- fragment loaders (wave32 WMMA, this code's contiguous k order shared by A and B, not the ISA table order) ----

// A-matrix f16 16x32: lane m = lane&15 holds row M=m; VGPR v holds K pair:
//   kb = 2v + (lane>=16 ? 16 : 0)
__device__ __forceinline__ v16h load_a_f16(const f16* base, int row0, int ld, int k0, int lane) {
    int m  = lane & 15;
    int hi = (lane >> 4) & 1;
    const f16* rp = base + (row0 + m) * ld + k0;
    v16h a;
#pragma unroll
    for (int v = 0; v < 8; ++v) {
        int kb = 2 * v + hi * 16;
        h2 p = *(const h2*)(rp + kb);
        a[2 * v] = p.x; a[2 * v + 1] = p.y;
    }
    return a;
}

// B-matrix f16 32x16 where element (K,N) = base[(col0+N)*ld + K]  (i.e. B = rows^T)
// lane n = lane&15 holds column N=n; VGPR v holds K pair kb = (lane>=16?16:0) + 2v
__device__ __forceinline__ v16h load_b_trans(const f16* base, int col0, int ld, int lane) {
    int n  = lane & 15;
    int hi = (lane >> 4) & 1;
    const f16* cp = base + (col0 + n) * ld + hi * 16;
    v16h b;
#pragma unroll
    for (int v = 0; v < 8; ++v) {
        h2 p = *(const h2*)(cp + 2 * v);
        b[2 * v] = p.x; b[2 * v + 1] = p.y;
    }
    return b;
}

// B-matrix f16 32x16 where element (K,N) = base[(k0+K)*ld + col0 + N]  (row-major V tile)
__device__ __forceinline__ v16h load_b_rows(const f16* base, int k0, int ld, int col0, int lane) {
    int n  = lane & 15;
    int hi = (lane >> 4) & 1;
    const f16* p = base + (k0 + hi * 16) * ld + col0 + n;
    v16h b;
#pragma unroll
    for (int v = 0; v < 8; ++v) {
        b[2 * v]     = p[(2 * v) * ld];
        b[2 * v + 1] = p[(2 * v + 1) * ld];
    }
    return b;
}

// One block = one (window, head) attention problem: S=256 tokens, hd=32.
extern "C" __global__ __launch_bounds__(256, 1)
void lepe_attn_kernel(const float* __restrict__ qkv,
                      const float* __restrict__ conv_w,   // (256,1,3,3) depthwise
                      const float* __restrict__ conv_b,   // (256,)
                      float* __restrict__ out) {
    extern __shared__ char smem[];
    f16* qs = (f16*)smem;                 // 256 x 32 f16, pre-scaled by 1/sqrt(hd)
    f16* ks = qs + SWIN * HD;             // 256 x 32 f16
    f16* vs = ks + SWIN * HD;             // 256 x 32 f16  (token-major, for WMMA B)
    f16* sh = vs + SWIN * HD;             // 256 x 264 f16 scores -> probs
    f16* vt = sh + SWIN * SH_LD;          // 32 x 256 f16  (channel-major, for LePE)

    const int tid  = threadIdx.x;
    const int lane = tid & 31;
    const int wave = tid >> 5;

    const int head = blockIdx.x & 7;
    const int win  = blockIdx.x >> 3;       // b*128 + d*4 + iW
    const int b    = win >> 7;
    const int d    = (win >> 2) & 31;
    const int iW   = win & 3;
    const int c0   = head * HD;

    // ---- stage q,k,v window tiles into LDS as f16 (coalesced float4 reads) ----
    const float* qg = qkv;
    const float* kg = qkv + (size_t)BATCH * LTOK * DIM;
    const float* vg = qkv + (size_t)2 * BATCH * LTOK * DIM;
    const int sub = tid & 7;                // which float4 within a 32-elem row
    const int rb  = tid >> 3;               // row within a group of 32 rows
#pragma unroll
    for (int it = 0; it < 8; ++it) {
        int s = it * 32 + rb;
        int h = s >> 3, w = s & 7;
        size_t l   = (size_t)d * 1024 + (size_t)h * 32 + iW * 8 + w;
        size_t off = ((size_t)b * LTOK + l) * DIM + c0 + sub * 4;
        f4 q4 = *(const f4*)(qg + off);
        f4 k4 = *(const f4*)(kg + off);
        f4 v4 = *(const f4*)(vg + off);
        int li = s * HD + sub * 4;
        qs[li + 0] = (f16)(q4.x * SCALE); qs[li + 1] = (f16)(q4.y * SCALE);
        qs[li + 2] = (f16)(q4.z * SCALE); qs[li + 3] = (f16)(q4.w * SCALE);
        ks[li + 0] = (f16)k4.x; ks[li + 1] = (f16)k4.y;
        ks[li + 2] = (f16)k4.z; ks[li + 3] = (f16)k4.w;
        f16 vh0 = (f16)v4.x, vh1 = (f16)v4.y, vh2 = (f16)v4.z, vh3 = (f16)v4.w;
        vs[li + 0] = vh0; vs[li + 1] = vh1; vs[li + 2] = vh2; vs[li + 3] = vh3;
        int e4 = sub * 4;                   // transposed copy for the LePE conv
        vt[(e4 + 0) * SWIN + s] = vh0;
        vt[(e4 + 1) * SWIN + s] = vh1;
        vt[(e4 + 2) * SWIN + s] = vh2;
        vt[(e4 + 3) * SWIN + s] = vh3;
    }
    __syncthreads();

    const int m  = lane & 15;
    const int hi = (lane >> 4) & 1;

    // ---- scores = (q*scale) @ k^T ----
    // Wave's 32 tiles have j in {wave, wave+8}: hoist both B fragments,
    // share each A fragment between the two j's.
    {
        v16h blo = load_b_trans(ks, wave * 16, HD, lane);
        v16h bhi = load_b_trans(ks, (wave + 8) * 16, HD, lane);
        for (int i = 0; i < 16; ++i) {
            v16h a = load_a_f16(qs, i * 16, HD, 0, lane);
            v8f acc0 = {}, acc1 = {};
            acc0 = __builtin_amdgcn_wmma_f32_16x16x32_f16(false, a, false, blo,
                                                          (short)0, acc0, false, false);
            acc1 = __builtin_amdgcn_wmma_f32_16x16x32_f16(false, a, false, bhi,
                                                          (short)0, acc1, false, false);
            f16* srow = sh + (i * 16 + hi * 8) * SH_LD + m;
#pragma unroll
            for (int r = 0; r < 8; ++r) {
                srow[r * SH_LD + wave * 16]       = (f16)acc0[r];
                srow[r * SH_LD + (wave + 8) * 16] = (f16)acc1[r];
            }
        }
    }
    __syncthreads();

    // ---- softmax: one f16 row per thread, streamed as 16-byte h8 vectors ----
    {
        h8* row = (h8*)(sh + tid * SH_LD);   // 32 x h8 = 256 scores
        h8 hm = row[0];
#pragma unroll 4
        for (int i2 = 1; i2 < 32; ++i2)
            hm = __builtin_elementwise_max(hm, row[i2]);   // v_pk_max f16
        float mx = (float)hm[0];
#pragma unroll
        for (int j2 = 1; j2 < 8; ++j2) {
            float x = (float)hm[j2];
            mx = x > mx ? x : mx;
        }
        float sum = 0.f;
        for (int i2 = 0; i2 < 32; ++i2) {
            h8 x = row[i2], y;
#pragma unroll
            for (int j2 = 0; j2 < 8; ++j2) {
                float e = __expf((float)x[j2] - mx);       // v_exp_f32 (TRANS)
                sum += e;
                y[j2] = (f16)e;
            }
            row[i2] = y;
        }
        f16 iv = (f16)(1.0f / sum);
        h8 inv8 = {iv, iv, iv, iv, iv, iv, iv, iv};
#pragma unroll 4
        for (int i2 = 0; i2 < 32; ++i2)
            row[i2] = row[i2] * inv8;                      // v_pk_mul_f16
    }
    __syncthreads();

    // ---- x = P @ V : each wave owns one e0 column and 4 row tiles ----
    // V B-fragment (kt) shared across the 4 tiles; 4 independent accumulators.
    const int e0 = (wave & 1) * 16;          // per-wave constant
    const int i0 = wave >> 1;                // tiles i0, i0+4, i0+8, i0+12
    const int e  = e0 + m;                   // head-dim index (fixed per lane)
    const int c  = c0 + e;                   // global channel (fixed per lane)

    float wgt[9];                            // conv weights: loaded once, overlap WMMAs
#pragma unroll
    for (int k9 = 0; k9 < 9; ++k9) wgt[k9] = conv_w[c * 9 + k9];
    float bias = conv_b[c];

    v8f acc[4] = {};
#pragma unroll
    for (int kt = 0; kt < 8; ++kt) {
        v16h bv = load_b_rows(vs, kt * 32, HD, e0, lane);
#pragma unroll
        for (int u = 0; u < 4; ++u) {
            v16h a = load_a_f16(sh, (i0 + 4 * u) * 16, SH_LD, kt * 32, lane);
            acc[u] = __builtin_amdgcn_wmma_f32_16x16x32_f16(false, a, false, bv,
                                                            (short)0, acc[u], false, false);
        }
    }

    // ---- epilogue: per lane each tile covers ONE image row h, wcol = 0..7 ----
#pragma unroll
    for (int u = 0; u < 4; ++u) {
        int i = i0 + 4 * u;
        int h = i * 2 + hi;                  // image row for all 8 acc elements
        float lep[8];
#pragma unroll
        for (int j = 0; j < 8; ++j) lep[j] = bias;
#pragma unroll
        for (int dy = -1; dy <= 1; ++dy) {
            int hh = h + dy;
            if (hh < 0 || hh >= 32) continue;
            h8 r8 = *(const h8*)(vt + e * SWIN + hh * 8);  // one aligned b128 load
            float f[10];
            f[0] = 0.f; f[9] = 0.f;
#pragma unroll
            for (int j = 0; j < 8; ++j) f[j + 1] = (float)r8[j];
            const float w0 = wgt[(dy + 1) * 3 + 0];
            const float w1 = wgt[(dy + 1) * 3 + 1];
            const float w2 = wgt[(dy + 1) * 3 + 2];
#pragma unroll
            for (int j = 0; j < 8; ++j)
                lep[j] += w0 * f[j] + w1 * f[j + 1] + w2 * f[j + 2];
        }
        size_t lrow = (size_t)d * 1024 + (size_t)h * 32 + iW * 8;
        float* orow = out + ((size_t)b * LTOK + lrow) * DIM + c;
#pragma unroll
        for (int r = 0; r < 8; ++r)          // wcol == r for this tile/lane
            orow[(size_t)r * DIM] = acc[u][r] + lep[r];
    }
}

extern "C" void kernel_launch(void* const* d_in, const int* in_sizes, int n_in,
                              void* d_out, int out_size, void* d_ws, size_t ws_size,
                              hipStream_t stream) {
    const float* qkv = (const float*)d_in[0];
    const float* cw  = (const float*)d_in[1];
    const float* cb  = (const float*)d_in[2];
    float* out = (float*)d_out;

    size_t shmem = (size_t)(3 * SWIN * HD) * sizeof(f16)    // 48 KB f16 q/k/v
                 + (size_t)SWIN * SH_LD * sizeof(f16)       // 132 KB f16 scores
                 + (size_t)HD * SWIN * sizeof(f16);         // 16 KB transposed V
    hipFuncSetAttribute((const void*)lepe_attn_kernel,
                        hipFuncAttributeMaxDynamicSharedMemorySize, (int)shmem);

    dim3 grid(BATCH * RES * 4 * HEADS);   // 2048 (window, head) problems
    dim3 block(256);                      // 8 waves
    lepe_attn_kernel<<<grid, block, shmem, stream>>>(qkv, cw, cb, out);
}
